// Multihead_self_attention_49675591745692
// MI455X (gfx1250) — matrix operand order repaired and checked by reading, not compiled, not run on hardware
//
#include <hip/hip_runtime.h>
#include <hip/hip_bf16.h>
#include <math.h>

typedef __bf16 bf16_t;
typedef __attribute__((ext_vector_type(16))) __bf16 v16bf;
typedef __attribute__((ext_vector_type(8)))  __bf16 v8bf;
typedef __attribute__((ext_vector_type(8)))  float  v8f;
typedef __attribute__((ext_vector_type(4)))  unsigned int u32x4;
typedef __attribute__((ext_vector_type(8)))  int i32x8;
typedef __attribute__((ext_vector_type(4)))  int i32x4;

#define WMMA_BF16(a, b, c) \
  __builtin_amdgcn_wmma_f32_16x16x32_bf16(false, (a), false, (b), (short)0, (c), false, false)

union AFragU { v16bf v; v8bf h[2]; };

// A-fragment (16x32 bf16, M x K), contiguous K order:
// lanes 0-15: row M=lane, K = k0+0..15; lanes 16-31: row M=lane-16, K = k0+16..31
// caller passes row pointer already offset by (m*ld + k0); half = lane>>4
template <typename P>
static __device__ inline v16bf load_afrag(const P* rowk, int half) {
  AFragU f;
  f.h[0] = *(const v8bf*)(rowk + half * 16);
  f.h[1] = *(const v8bf*)(rowk + half * 16 + 8);
  return f.v;
}

// ---------------------------------------------------------------------------
// TDM: issue a 2-D tensor tile load Global -> LDS (bf16 elements).
// Descriptor packing per CDNA5 ISA.
//   group0: count=1 | lds_addr | global_addr | type=2
//   group1: data_size=1(2B) | tensor dims | tile dims | dim0 stride
// Issued uniformly (EXEC ignored by TDM); tracked with TENSORcnt.
// This toolchain exposes the 6-arg builtin (g0, g1, g2, g3, g4, cpol).
// ---------------------------------------------------------------------------
static __device__ inline void tdm_load_tile_bf16(const bf16_t* gaddr, unsigned lds_byte_off,
                                                 unsigned tile_k, unsigned tile_rows,
                                                 unsigned tensor_k, unsigned tensor_rows,
                                                 unsigned row_stride_elems) {
  unsigned long long ga = (unsigned long long)(uintptr_t)gaddr;
  u32x4 g0;
  g0[0] = 1u;                                              // count=1 (valid), user mode
  g0[1] = lds_byte_off;                                    // lds_addr (bytes)
  g0[2] = (unsigned)(ga & 0xffffffffu);                    // global_addr[31:0]
  g0[3] = (unsigned)((ga >> 32) & 0x01ffffffu) | (2u << 30);  // global_addr[56:32] | type=2
  i32x8 g1;
  g1[0] = (int)(1u << 16);                                 // workgroup_mask=0, data_size=1 (2 bytes)
  g1[1] = (int)((tensor_k & 0xffffu) << 16);               // tensor_dim0[15:0]
  g1[2] = (int)((tensor_k >> 16) | ((tensor_rows & 0xffffu) << 16));  // dim0[31:16] | dim1[15:0]
  g1[3] = (int)((tensor_rows >> 16) | (tile_k << 16));     // dim1[31:16] | tile_dim0
  g1[4] = (int)tile_rows;                                  // tile_dim1 (tile_dim2=0)
  g1[5] = (int)row_stride_elems;                           // tensor_dim0_stride[31:0]
  g1[6] = 0;                                               // stride0[47:32] | stride1[15:0]
  g1[7] = 0;
  i32x4 z4 = {0, 0, 0, 0};
  i32x8 z8 = {0, 0, 0, 0, 0, 0, 0, 0};
  __builtin_amdgcn_tensor_load_to_lds(g0, g1, z4, z4, z8, 0);
}

// ---------------------------------------------------------------------------
// fp32 -> bf16 conversion
// ---------------------------------------------------------------------------
__global__ void cvt_f32_bf16(const float* __restrict__ in, bf16_t* __restrict__ out, int n) {
  int i = blockIdx.x * blockDim.x + threadIdx.x;
  int stride = gridDim.x * blockDim.x;
  for (; i < n; i += stride) out[i] = (bf16_t)in[i];
}

// ---------------------------------------------------------------------------
// LDS-tiled GEMM with TDM double-buffering.
//   out[m,n] = sum_k A[m,k] * W[n,k]   A: [8192,1024] bf16, W: [1024,1024] bf16 row-major [n][k]
// Block = 256 threads (8 waves), block tile = 128(M) x 64(N), K-step 32.
// Wave 0 drives the Tensor Data Mover: per K-step it stages A(128x32) and B(64x32)
// slices into LDS (double buffered), waits on TENSORcnt, then all 8 waves consume
// the staged tiles via ds loads feeding v_wmma (B shared by all 8 waves; each wave reads its own 16 A rows).
// mode 0: Q/K layout [b,h,s,d] bf16 ; mode 1: V^T layout [b,h,d,s] bf16 ; mode 2: fp32 row-major
// grid = (8192/128)*(1024/64) = 1024 blocks.
// ---------------------------------------------------------------------------
__global__ void gemm_tdm(const bf16_t* __restrict__ A,
                         const bf16_t* __restrict__ W,
                         void* __restrict__ outp, int mode) {
  __shared__ __align__(16) bf16_t sA[2][128 * 32];   // 2 x 8 KiB
  __shared__ __align__(16) bf16_t sB[2][64 * 32];    // 2 x 4 KiB

  const int lane = threadIdx.x & 31;
  const int wave = threadIdx.x >> 5;                 // 0..7
  const int mtile = blockIdx.x >> 4;                 // 0..63
  const int ntile = blockIdx.x & 15;                 // 0..15 (== head for mode 0/1)
  const int m0 = mtile * 128;
  const int n0 = ntile * 64;
  const int mw = m0 + wave * 16;                     // this wave's 16 output rows
  const int half = lane >> 4, nl = lane & 15;
  const bool issuer = (wave == 0);

  const bf16_t* Abase = A + (size_t)m0 * 1024;
  const bf16_t* Wbase = W + (size_t)n0 * 1024;

  v8f c[4] = {};

  if (issuer) {
    tdm_load_tile_bf16(Abase, (unsigned)(uintptr_t)&sA[0][0], 32, 128, 1024, 8192, 1024);
    tdm_load_tile_bf16(Wbase, (unsigned)(uintptr_t)&sB[0][0], 32, 64, 1024, 1024, 1024);
  }

  const int NK = 32;
  for (int kb = 0; kb < NK; ++kb) {
    const int buf = kb & 1;
    if (issuer) {
      if (kb + 1 < NK) {
        tdm_load_tile_bf16(Abase + (kb + 1) * 32, (unsigned)(uintptr_t)&sA[buf ^ 1][0],
                           32, 128, 1024, 8192, 1024);
        tdm_load_tile_bf16(Wbase + (kb + 1) * 32, (unsigned)(uintptr_t)&sB[buf ^ 1][0],
                           32, 64, 1024, 1024, 1024);
        __builtin_amdgcn_s_wait_tensorcnt(2);   // pair kb complete, pair kb+1 in flight
      } else {
        __builtin_amdgcn_s_wait_tensorcnt(0);
      }
    }
    __syncthreads();

    v16bf a = load_afrag(&sA[buf][(size_t)((wave * 16) + nl) * 32], half);
#pragma unroll
    for (int t = 0; t < 4; ++t) {
      v16bf b = *(const v16bf*)&sB[buf][(size_t)(t * 16 + nl) * 32 + half * 16];
      c[t] = WMMA_BF16(a, b, c[t]);
    }
    __syncthreads();   // protect buf from being overwritten by the kb+2 issue
  }

  // ---- store
  const int b_idx = mw >> 11;                        // 128-row blocks never straddle batch
  if (mode == 2) {
    float* out = (float*)outp;
#pragma unroll
    for (int t = 0; t < 4; ++t) {
#pragma unroll
      for (int r = 0; r < 8; ++r) {
        int m = mw + r + 8 * half;
        out[(size_t)m * 1024 + n0 + t * 16 + nl] = c[t][r];
      }
    }
  } else {
    bf16_t* out = (bf16_t*)outp;
#pragma unroll
    for (int t = 0; t < 4; ++t) {
#pragma unroll
      for (int r = 0; r < 8; ++r) {
        int m = mw + r + 8 * half;
        int s = m & 2047;
        int d = t * 16 + nl;
        size_t idx = (mode == 0)
                         ? (((size_t)(b_idx * 16 + ntile) * 2048 + s) * 64 + d)
                         : (((size_t)(b_idx * 16 + ntile) * 64 + d) * 2048 + s);
        out[idx] = (bf16_t)c[t][r];
      }
    }
  }
}

// ---------------------------------------------------------------------------
// Flash attention, causal. One wave per (b, h, 16-row q-tile).
// Q,K: [b*h, 2048, 64] bf16 ; Vt: [b*h, 64, 2048] bf16 ; heads: [b, 2048, 1024] bf16
// blockDim=128 (4 waves), grid=2048 -> 8192 waves.
// ---------------------------------------------------------------------------
__global__ void flash_attn(const bf16_t* __restrict__ Q,
                           const bf16_t* __restrict__ K,
                           const bf16_t* __restrict__ Vt,
                           bf16_t* __restrict__ heads) {
  __shared__ __align__(32) bf16_t Pl[4][16 * 32];

  const int lane = threadIdx.x & 31;
  const int wib  = threadIdx.x >> 5;
  const int wid  = blockIdx.x * 4 + wib;
  const int bh = wid >> 7;     // 0..63 = b*16+h
  const int qt = wid & 127;
  const int q0 = qt * 16;
  const int half = lane >> 4, nl = lane & 15;

  const bf16_t* Qh = Q + ((size_t)bh * 2048 + q0) * 64;
  const bf16_t* Kh = K + (size_t)bh * 2048 * 64;
  const bf16_t* Vh = Vt + (size_t)bh * 64 * 2048;

  v16bf qa0 = load_afrag(Qh + nl * 64, half);
  v16bf qa1 = load_afrag(Qh + nl * 64 + 32, half);

  float mrow[8], lrow[8];
  v8f o[4] = {};
#pragma unroll
  for (int r = 0; r < 8; ++r) { mrow[r] = -3.0e38f; lrow[r] = 0.f; }

  const float scale = 0.125f;  // 1/sqrt(64)
  const int nkb = (q0 + 16 + 31) >> 5;

  for (int kb = 0; kb < nkb; ++kb) {
    const int k0 = kb * 32;

    // ---- S = Q K^T for two 16-key column tiles
    v8f s0 = {}, s1 = {};
    {
      const bf16_t* kr0 = Kh + (size_t)(k0 + nl) * 64 + half * 16;
      v16bf b00 = *(const v16bf*)(kr0);
      v16bf b01 = *(const v16bf*)(kr0 + 32);
      s0 = WMMA_BF16(qa0, b00, s0);
      s0 = WMMA_BF16(qa1, b01, s0);
      const bf16_t* kr1 = Kh + (size_t)(k0 + 16 + nl) * 64 + half * 16;
      v16bf b10 = *(const v16bf*)(kr1);
      v16bf b11 = *(const v16bf*)(kr1 + 32);
      s1 = WMMA_BF16(qa0, b10, s1);
      s1 = WMMA_BF16(qa1, b11, s1);
    }

    // ---- causal mask + online softmax (rows live across 16-lane halves)
    const int kc0 = k0 + nl, kc1 = k0 + 16 + nl;
    float p0[8], p1[8], alpha[8];
#pragma unroll
    for (int r = 0; r < 8; ++r) {
      int q = q0 + r + 8 * half;
      float v0 = (kc0 <= q) ? s0[r] * scale : -3.0e38f;
      float v1 = (kc1 <= q) ? s1[r] * scale : -3.0e38f;
      float t = fmaxf(v0, v1);
      t = fmaxf(t, __shfl_xor(t, 1, 32));
      t = fmaxf(t, __shfl_xor(t, 2, 32));
      t = fmaxf(t, __shfl_xor(t, 4, 32));
      t = fmaxf(t, __shfl_xor(t, 8, 32));
      float mn = fmaxf(mrow[r], t);
      alpha[r] = __expf(mrow[r] - mn);
      mrow[r] = mn;
      p0[r] = __expf(v0 - mn);
      p1[r] = __expf(v1 - mn);
      float rs = p0[r] + p1[r];
      rs += __shfl_xor(rs, 1, 32);
      rs += __shfl_xor(rs, 2, 32);
      rs += __shfl_xor(rs, 4, 32);
      rs += __shfl_xor(rs, 8, 32);
      lrow[r] = lrow[r] * alpha[r] + rs;
    }
#pragma unroll
    for (int t = 0; t < 4; ++t) {
#pragma unroll
      for (int r = 0; r < 8; ++r) o[t][r] = o[t][r] * alpha[r];
    }

    // ---- restripe P from C-layout to A-layout via LDS (16x32 bf16 row-major)
#pragma unroll
    for (int r = 0; r < 8; ++r) {
      int row = r + 8 * half;
      Pl[wib][row * 32 + nl]      = (bf16_t)p0[r];
      Pl[wib][row * 32 + 16 + nl] = (bf16_t)p1[r];
    }
    asm volatile("s_wait_dscnt 0" ::: "memory");
    AFragU pf;
    pf.h[0] = *(const v8bf*)&Pl[wib][nl * 32 + half * 16];
    pf.h[1] = *(const v8bf*)&Pl[wib][nl * 32 + half * 16 + 8];
    v16bf pa = pf.v;

    // ---- O += P * V  (V^T layout makes per-lane B loads contiguous)
#pragma unroll
    for (int t = 0; t < 4; ++t) {
      v16bf bv = *(const v16bf*)(Vh + (size_t)(t * 16 + nl) * 2048 + k0 + half * 16);
      o[t] = WMMA_BF16(pa, bv, o[t]);
    }
  }

  // ---- normalize and write heads in [b, s, h*64+d] (row-major for final GEMM)
  const int b_idx = bh >> 4, h = bh & 15;
#pragma unroll
  for (int t = 0; t < 4; ++t) {
#pragma unroll
    for (int r = 0; r < 8; ++r) {
      int q = q0 + r + 8 * half;
      heads[((size_t)b_idx * 2048 + q) * 1024 + h * 64 + t * 16 + nl] =
          (bf16_t)(o[t][r] / lrow[r]);
    }
  }
}

// ---------------------------------------------------------------------------
extern "C" void kernel_launch(void* const* d_in, const int* in_sizes, int n_in,
                              void* d_out, int out_size, void* d_ws, size_t ws_size,
                              hipStream_t stream) {
  (void)in_sizes; (void)n_in; (void)out_size; (void)ws_size;
  const float* x  = (const float*)d_in[0];
  const float* Wq = (const float*)d_in[1];
  const float* Wk = (const float*)d_in[2];
  const float* Wv = (const float*)d_in[3];
  const float* Wo = (const float*)d_in[4];
  float* out = (float*)d_out;

  const size_t M = 8192, DM = 1024;
  char* ws = (char*)d_ws;
  bf16_t* xb  = (bf16_t*)ws; ws += M * DM * 2;     // 16 MiB
  bf16_t* wqb = (bf16_t*)ws; ws += DM * DM * 2;    //  2 MiB
  bf16_t* wkb = (bf16_t*)ws; ws += DM * DM * 2;
  bf16_t* wvb = (bf16_t*)ws; ws += DM * DM * 2;
  bf16_t* wob = (bf16_t*)ws; ws += DM * DM * 2;
  bf16_t* Qb  = (bf16_t*)ws; ws += M * DM * 2;     // [b,h,s,d]
  bf16_t* Kb  = (bf16_t*)ws; ws += M * DM * 2;     // [b,h,s,d]
  bf16_t* Vtb = (bf16_t*)ws; ws += M * DM * 2;     // [b,h,d,s]
  bf16_t* hb  = (bf16_t*)ws; ws += M * DM * 2;     // [b,s,1024]

  cvt_f32_bf16<<<2048, 256, 0, stream>>>(x, xb, (int)(M * DM));
  cvt_f32_bf16<<<512, 256, 0, stream>>>(Wq, wqb, (int)(DM * DM));
  cvt_f32_bf16<<<512, 256, 0, stream>>>(Wk, wkb, (int)(DM * DM));
  cvt_f32_bf16<<<512, 256, 0, stream>>>(Wv, wvb, (int)(DM * DM));
  cvt_f32_bf16<<<512, 256, 0, stream>>>(Wo, wob, (int)(DM * DM));

  gemm_tdm<<<1024, 256, 0, stream>>>(xb, wqb, (void*)Qb, 0);
  gemm_tdm<<<1024, 256, 0, stream>>>(xb, wkb, (void*)Kb, 0);
  gemm_tdm<<<1024, 256, 0, stream>>>(xb, wvb, (void*)Vtb, 1);

  flash_attn<<<2048, 128, 0, stream>>>(Qb, Kb, Vtb, hb);

  gemm_tdm<<<1024, 256, 0, stream>>>(hb, wob, (void*)out, 2);
}
